// DeepDFA_19851338842260
// MI455X (gfx1250) — matrix operand order repaired and checked by reading, not compiled, not run on hardware
//
#include <hip/hip_runtime.h>

typedef __attribute__((ext_vector_type(16))) _Float16 v16h;
typedef __attribute__((ext_vector_type(8)))  float    v8f;

#define NS      64
#define NA      32
#define BATCH   4096
#define SEQLEN  512
#define REWARD_FLOATS (BATCH * SEQLEN * 2)   // 4194304 floats

// ---------------------------------------------------------------------------
// Kernel 1: distill the integer next-state table from the one-hot float
// transition tensor via WMMA:   next[r] = sum_k trans_prob[r][k] * k
// (r = a*64+s flattened -> 2048 rows x 64 cols).
// Formulated as D = A x B with A(16x32) = ramp (A[m][k] = k for all m) and
// B(32x16) = P^T tile (B[k][n] = P[r0+n][k]), accumulated over K with two
// v_wmma_f32_16x16x32_f16. Every row of D is identical, so each lane's c[0]
// holds next[r0 + lane%16] (only the C/D N=lane%16 mapping
// is needed to scatter the result). All arithmetic is exact (0/1 x small int).
// ---------------------------------------------------------------------------
__global__ void dfa_extract_table(const float* __restrict__ trans_prob,
                                  unsigned char* __restrict__ ns_out) {
    const int tile = blockIdx.x;            // 128 tiles of 16 rows
    const int lane = threadIdx.x;           // one wave32 per block
    const int r0   = tile * 16;
    const int col  = lane & 15;             // B column fed by this lane
    const int kbaseA = (lane < 16) ? 0 : 8;   // 16-bit A layout: K split by lane half
    const int kbaseB = (lane < 16) ? 0 : 16;  // contiguous K order: lane half h holds k = 16h + e

    const float* Prow = trans_prob + (size_t)(r0 + col) * NS;

    v16h a1, a2, b1, b2;
#pragma unroll
    for (int e = 0; e < 16; ++e) {
        int kA = kbaseB + e;                        // K index of element e in A
        a1[e] = (_Float16)(float)kA;
        a2[e] = (_Float16)(float)(kA + 32);
        int kB = kbaseB + e;                        // K index of element e in B
        b1[e] = (_Float16)Prow[kB];
        b2[e] = (_Float16)Prow[kB + 32];
    }

    v8f c = {};
    c = __builtin_amdgcn_wmma_f32_16x16x32_f16(false, a1, false, b1,
                                               (short)0, c, false, false);
    c = __builtin_amdgcn_wmma_f32_16x16x32_f16(false, a2, false, b2,
                                               (short)0, c, false, false);

    if (lane < 16) {
        int v = (int)(c[0] + 0.5f);
        v = v < 0 ? 0 : (v > 63 ? 63 : v);
        ns_out[r0 + lane] = (unsigned char)v;
    }
}

// ---------------------------------------------------------------------------
// Kernel 2: the DFA walk. One thread per batch row. A 64KB two-step table
// T2[a1][a2][s] = T[a2][T[a1][s]] in LDS (only possible with CDNA5's 320KB
// WGP LDS) halves the dependent-load chain to 256 steps; the intermediate
// state and both accepting-row lookups are off the critical path.
// ---------------------------------------------------------------------------
__global__ void __launch_bounds__(128)
dfa_walk(const int* __restrict__ action_seq,
         const float* __restrict__ accepting,
         const unsigned char* __restrict__ ns_tab,
         float* __restrict__ out) {
    __shared__ __align__(16) unsigned char nsL[NA * NS];        // 2 KB
    __shared__ __align__(16) unsigned char t2L[NA * NA * NS];   // 64 KB
    __shared__ float2 accL[NS];                                 // 0.5 KB

    const int tid = threadIdx.x;

    // Stage single-step table (2048 B) and accepting rows into LDS.
    ((int4*)nsL)[tid] = ((const int4*)ns_tab)[tid];             // 128 x 16 B
    if (tid < NS) accL[tid] = ((const float2*)accepting)[tid];
    __syncthreads();

    // Build the two-step table: 512 independent (pipelined) entries/thread.
    for (int i = tid; i < NA * NA * NS; i += 128) {
        int a1 = i >> 11, a2 = (i >> 6) & 31, s = i & 63;
        int t1 = nsL[(a1 << 6) | s];
        t2L[i] = nsL[(a2 << 6) | t1];
    }
    __syncthreads();

    const int b = blockIdx.x * 128 + tid;                       // 32 blocks
    const int* act = action_seq + (size_t)b * SEQLEN;
    float4* rw = (float4*)(out + (size_t)b * (SEQLEN * 2));
    int s = 0;                                                  // s0 = one-hot(0)

#pragma unroll 1
    for (int mi = 0; mi < 32; ++mi) {                           // 16 actions / iter
        const int4* ap = (const int4*)(act + mi * 16);
        int4 w0 = ap[0], w1 = ap[1], w2 = ap[2], w3 = ap[3];
        if (mi < 31) __builtin_prefetch(act + (mi + 1) * 16, 0, 0);
        int as[16] = {w0.x, w0.y, w0.z, w0.w,  w1.x, w1.y, w1.z, w1.w,
                      w2.x, w2.y, w2.z, w2.w,  w3.x, w3.y, w3.z, w3.w};
#pragma unroll
        for (int p = 0; p < 8; ++p) {
            int a1 = as[2 * p], a2 = as[2 * p + 1];
            int mid = nsL[(a1 << 6) | s];                       // state after a1
            s = t2L[(a1 << 11) | (a2 << 6) | s];                // chain: 1 ds_load_u8
            float2 r0 = accL[mid];
            float2 r1 = accL[s];
            rw[mi * 8 + p] = make_float4(r0.x, r0.y, r1.x, r1.y);
        }
    }

    // One-hot final state, branch-free float4 stores.
    float* sf = out + (size_t)REWARD_FLOATS + (size_t)b * NS;
#pragma unroll
    for (int j = 0; j < 16; ++j) {
        int base = j * 4;
        float4 v;
        v.x = (s == base + 0) ? 1.0f : 0.0f;
        v.y = (s == base + 1) ? 1.0f : 0.0f;
        v.z = (s == base + 2) ? 1.0f : 0.0f;
        v.w = (s == base + 3) ? 1.0f : 0.0f;
        ((float4*)sf)[j] = v;
    }
}

// ---------------------------------------------------------------------------
extern "C" void kernel_launch(void* const* d_in, const int* in_sizes, int n_in,
                              void* d_out, int out_size, void* d_ws, size_t ws_size,
                              hipStream_t stream) {
    const int*   action_seq = (const int*)d_in[0];    // [4096, 512] int32
    const float* trans_prob = (const float*)d_in[1];  // [32, 64, 64] f32 one-hot
    const float* accepting  = (const float*)d_in[2];  // [64, 2] f32
    unsigned char* ns_tab   = (unsigned char*)d_ws;   // 2 KB scratch table

    dfa_extract_table<<<dim3(128), dim3(32), 0, stream>>>(trans_prob, ns_tab);
    dfa_walk<<<dim3(32), dim3(128), 0, stream>>>(action_seq, accepting, ns_tab,
                                                 (float*)d_out);
}
